// HGT_781684048102
// MI455X (gfx1250) — matrix operand order repaired and checked by reading, not compiled, not run on hardware
//
#include <hip/hip_runtime.h>
#include <math.h>

// ---------------------------------------------------------------------------
// Types for WMMA
// ---------------------------------------------------------------------------
typedef _Float16 v16h __attribute__((ext_vector_type(16)));
typedef _Float16 v8h  __attribute__((ext_vector_type(8)));
typedef float    v8f  __attribute__((ext_vector_type(8)));

#define NN_NODES 100000
#define NE_EDGES 500000
#define HID 128
#define NH 8
#define HD 16

// ---------------------------------------------------------------------------
// Utility kernels
// ---------------------------------------------------------------------------
__global__ void fill_f32(float* __restrict__ p, float v, long n) {
    long t = (long)blockIdx.x * blockDim.x + threadIdx.x;
    if (t < n) p[t] = v;
}

__global__ void f32_to_f16(const float* __restrict__ x, _Float16* __restrict__ y, long n) {
    long t = (long)blockIdx.x * blockDim.x + threadIdx.x;
    if (t < n) y[t] = (_Float16)x[t];
}

// gelu(x) (exact) -> f16 buffer (A-operand of the a-projection GEMM)
__global__ void gelu_to_f16(const float* __restrict__ x, _Float16* __restrict__ y, long n) {
    long t = (long)blockIdx.x * blockDim.x + threadIdx.x;
    if (t < n) {
        float v = x[t];
        float g = 0.5f * v * (1.0f + erff(v * 0.70710678118654752f));
        y[t] = (_Float16)g;
    }
}

// w[K x Nout] f32 (row-major, as in reference x@W) -> wt[Nout x K] f16
__global__ void transpose_w_f16(const float* __restrict__ w, _Float16* __restrict__ wt,
                                int K, int Nout) {
    long t = (long)blockIdx.x * blockDim.x + threadIdx.x;
    if (t >= (long)K * Nout) return;
    int o = (int)(t / K);
    int k = (int)(t % K);
    wt[(size_t)o * K + k] = (_Float16)w[(size_t)k * Nout + o];
}

// Fold per-relation head transform into the projection:
//   fused[c][o=h*16+f] = sum_d w[c][h*16+d] * rel[h][d][f]   -> wt16[o][c]
//   bf[o]              = sum_d b[h*16+d]    * rel[h][d][f]
__global__ void fuse_rel_w(const float* __restrict__ w, const float* __restrict__ rel,
                           const float* __restrict__ b,
                           _Float16* __restrict__ wt, float* __restrict__ bf) {
    int t = blockIdx.x * blockDim.x + threadIdx.x;
    if (t >= HID * HID) return;
    int c = t >> 7;
    int o = t & 127;
    int h = o >> 4;
    int f = o & 15;
    float s = 0.0f;
#pragma unroll
    for (int d = 0; d < 16; ++d)
        s += w[c * HID + h * 16 + d] * rel[h * 256 + d * 16 + f];
    wt[(size_t)o * HID + c] = (_Float16)s;
    if (c == 0) {
        float sb = 0.0f;
#pragma unroll
        for (int d = 0; d < 16; ++d)
            sb += b[h * 16 + d] * rel[h * 256 + d * 16 + f];
        bf[o] = sb;
    }
}

// ---------------------------------------------------------------------------
// WMMA GEMM:  C[M x 128] = A[M x K](f16) * Bt^T  (Bt stored [128 x K] f16)
//             + bias, optional ReLU, optional learnable-sigmoid skip,
//             optional f16 mirror of the output.
// Block = 256 threads (8 waves); wave w owns output cols [16w,16w+16),
// block b owns rows [16b,16b+16). M must be a multiple of 16 (100000 is).
// Fragment packing (wave32), contiguous k = 16*(lane/16) + s for both A and B:
//   A 16x32 f16: lane -> M=lane%16,, halves: K = kb+{0..7} then kb+{8..15},
//                kb = 0 (lanes 0-15) / 16 (lanes 16-31) => two 16B loads.
//   B 32x16 f16: lane -> N=lane%16, halves: K = kOff+{0..15},
//                kOff = 0 / 16                            => one 32B load.
//   C 16x16 f32: lane -> N=lane%16, vgpr r -> M = r + (lane<16 ? 0 : 8).
// ---------------------------------------------------------------------------
__global__ __launch_bounds__(256) void gemm_wmma_f16(
    const _Float16* __restrict__ A, const _Float16* __restrict__ Bt,
    const float* __restrict__ bias, float* __restrict__ C32,
    _Float16* __restrict__ C16, const float* __restrict__ skipX,
    const float* __restrict__ skipP, int act, int K) {
    const int lane = threadIdx.x & 31;
    const int wave = threadIdx.x >> 5;
    const int row0 = blockIdx.x * 16;
    const int col0 = wave * 16;

    const int m  = lane & 15;
    const int kb = (lane < 16) ? 0 : 16;
    const int n  = lane & 15;
    const int kOffB = (lane < 16) ? 0 : 16;

    const _Float16* arow = A  + (size_t)(row0 + m) * K + kb;
    const _Float16* brow = Bt + (size_t)(col0 + n) * K + kOffB;

    v8f c = {};
    for (int kk = 0; kk < K; kk += 32) {
        v8h lo = *(const v8h*)(arow + kk);
        v8h hi = *(const v8h*)(arow + kk + 8);
        v16h a;
#pragma unroll
        for (int i = 0; i < 8; ++i) { a[i] = lo[i]; a[i + 8] = hi[i]; }
        v16h b = *(const v16h*)(brow + kk);
        c = __builtin_amdgcn_wmma_f32_16x16x32_f16(
                false, a, false, b, (short)0, c, false, false);
    }

    const int col = col0 + n;
    const float bcol = bias ? bias[col] : 0.0f;
    float sk = 0.0f, omsk = 0.0f;
    if (skipP) { sk = 1.0f / (1.0f + __expf(-skipP[0])); omsk = 1.0f - sk; }
#pragma unroll
    for (int r = 0; r < 8; ++r) {
        int row = row0 + r + ((lane < 16) ? 0 : 8);
        float v = c[r] + bcol;
        if (act == 1) v = fmaxf(v, 0.0f);
        size_t oidx = (size_t)row * HID + col;
        if (skipP) v = sk * v + omsk * skipX[oidx];
        if (C32) C32[oidx] = v;
        if (C16) C16[oidx] = (_Float16)v;
    }
}

// ---------------------------------------------------------------------------
// Edge-phase kernels (one wave32 per edge; lane owns float4 slice, head = lane/4)
// ---------------------------------------------------------------------------
__device__ __forceinline__ float atomicMaxF(float* a, float v) {
    return (v >= 0.0f)
        ? __int_as_float(atomicMax((int*)a, __float_as_int(v)))
        : __uint_as_float(atomicMin((unsigned int*)a, __float_as_uint(v)));
}

__global__ __launch_bounds__(256) void edge_alpha_max(
    const float* __restrict__ q, const float* __restrict__ krel,
    const int* __restrict__ ei, const float* __restrict__ prel,
    float* __restrict__ alpha, float* __restrict__ mbuf, int E) {
    int e = blockIdx.x * (blockDim.x >> 5) + (threadIdx.x >> 5);
    int lane = threadIdx.x & 31;
    if (e >= E) return;
    int si = ei[e], di = ei[E + e];
    float4 qv = ((const float4*)q)[(size_t)di * 32 + lane];
    float4 kv = ((const float4*)krel)[(size_t)si * 32 + lane];
    float p = qv.x * kv.x + qv.y * kv.y + qv.z * kv.z + qv.w * kv.w;
    p += __shfl_xor(p, 1, 32);
    p += __shfl_xor(p, 2, 32);          // all 4 lanes of a head now hold the dot
    int h = lane >> 2;
    float a = p * prel[h] * 0.25f;       // scale = 1/sqrt(D)
    if ((lane & 3) == 0) {
        alpha[(size_t)e * NH + h] = a;
        atomicMaxF(&mbuf[(size_t)di * NH + h], a);
    }
}

__global__ void edge_exp_sum(float* __restrict__ alpha, const float* __restrict__ mbuf,
                             float* __restrict__ sbuf, const int* __restrict__ ei, int E) {
    long t = (long)blockIdx.x * blockDim.x + threadIdx.x;
    if (t >= (long)E * NH) return;
    int e = (int)(t >> 3);
    int h = (int)(t & 7);
    int di = ei[E + e];
    float w = __expf(alpha[t] - mbuf[(size_t)di * NH + h]);
    alpha[t] = w;
    atomicAdd(&sbuf[(size_t)di * NH + h], w);
}

__global__ __launch_bounds__(256) void edge_scatter(
    const float* __restrict__ vrel, const float* __restrict__ alpha,
    const float* __restrict__ sbuf, const int* __restrict__ ei,
    float* __restrict__ agg, int E) {
    int e = blockIdx.x * (blockDim.x >> 5) + (threadIdx.x >> 5);
    int lane = threadIdx.x & 31;
    if (e >= E) return;
    int si = ei[e], di = ei[E + e];
    int h = lane >> 2;
    float w = alpha[(size_t)e * NH + h];
    float s = sbuf[(size_t)di * NH + h];
    float coef = w / (s + 1e-16f);
    float4 vv = ((const float4*)vrel)[(size_t)si * 32 + lane];
    float* dst = agg + (size_t)di * HID + lane * 4;
    atomicAdd(dst + 0, vv.x * coef);
    atomicAdd(dst + 1, vv.y * coef);
    atomicAdd(dst + 2, vv.z * coef);
    atomicAdd(dst + 3, vv.w * coef);
}

// Final 128 -> 1 projection (gmd nodes). One wave per node, full-wave reduce.
__global__ __launch_bounds__(256) void out_proj(
    const float* __restrict__ x, const float* __restrict__ w,
    const float* __restrict__ b, float* __restrict__ y, int N) {
    int node = blockIdx.x * (blockDim.x >> 5) + (threadIdx.x >> 5);
    int lane = threadIdx.x & 31;
    if (node >= N) return;
    float4 xv = ((const float4*)x)[(size_t)node * 32 + lane];
    float4 wv = ((const float4*)w)[lane];
    float p = xv.x * wv.x + xv.y * wv.y + xv.z * wv.z + xv.w * wv.w;
#pragma unroll
    for (int off = 16; off > 0; off >>= 1) p += __shfl_xor(p, off, 32);
    if (lane == 0) y[node] = p + b[0];
}

// ---------------------------------------------------------------------------
// Host orchestration
// ---------------------------------------------------------------------------
// Input index map (setup_inputs dict order, params flattened in insertion order)
#define IDX_INW(t)     (6 + 2 * (t))
#define IDX_INB(t)     (7 + 2 * (t))
#define LBASE(l)       (10 + 30 * (l))
#define IDX_KW(l, t)   (LBASE(l) + 9 * (t) + 0)
#define IDX_KB(l, t)   (LBASE(l) + 9 * (t) + 1)
#define IDX_QW(l, t)   (LBASE(l) + 9 * (t) + 2)
#define IDX_QB(l, t)   (LBASE(l) + 9 * (t) + 3)
#define IDX_VW(l, t)   (LBASE(l) + 9 * (t) + 4)
#define IDX_VB(l, t)   (LBASE(l) + 9 * (t) + 5)
#define IDX_AW(l, t)   (LBASE(l) + 9 * (t) + 6)
#define IDX_AB(l, t)   (LBASE(l) + 9 * (t) + 7)
#define IDX_SKIP(l, t) (LBASE(l) + 9 * (t) + 8)
#define IDX_AREL(l, r) (LBASE(l) + 18 + 3 * (r) + 0)
#define IDX_MREL(l, r) (LBASE(l) + 18 + 3 * (r) + 1)
#define IDX_PREL(l, r) (LBASE(l) + 18 + 3 * (r) + 2)
#define IDX_OUTW 70
#define IDX_OUTB 71

extern "C" void kernel_launch(void* const* d_in, const int* in_sizes, int n_in,
                              void* d_out, int out_size, void* d_ws, size_t ws_size,
                              hipStream_t stream) {
    (void)in_sizes; (void)n_in; (void)out_size; (void)ws_size;
    const int NN = NN_NODES;
    const int E  = NE_EDGES;

    const float* xin[2] = { (const float*)d_in[0], (const float*)d_in[1] };
    const int*   ei[4]  = { (const int*)d_in[2], (const int*)d_in[3],
                            (const int*)d_in[4], (const int*)d_in[5] };
    auto P = [&](int i) { return (const float*)d_in[i]; };

    // relation -> (src type, dst type):  bb gg bg gb
    const int rsrc[4] = { 0, 1, 0, 1 };
    const int rdst[4] = { 0, 1, 1, 0 };

    // ---- workspace carve-out ----
    size_t off = 0;
    auto alloc = [&](size_t bytes) {
        void* p = (char*)d_ws + off;
        off = (off + bytes + 255) & ~(size_t)255;
        return p;
    };
    _Float16* xin16[2]; float* xa32[2]; _Float16* xa16[2];
    float* xb32[2]; _Float16* xb16[2];
    float* qb[2]; float* agg[2];
    for (int t = 0; t < 2; ++t) xin16[t] = (_Float16*)alloc((size_t)NN * 64 * 2);
    for (int t = 0; t < 2; ++t) { xa32[t] = (float*)alloc((size_t)NN * HID * 4);
                                  xa16[t] = (_Float16*)alloc((size_t)NN * HID * 2); }
    for (int t = 0; t < 2; ++t) { xb32[t] = (float*)alloc((size_t)NN * HID * 4);
                                  xb16[t] = (_Float16*)alloc((size_t)NN * HID * 2); }
    for (int t = 0; t < 2; ++t) qb[t]  = (float*)alloc((size_t)NN * HID * 4);
    for (int t = 0; t < 2; ++t) agg[t] = (float*)alloc((size_t)NN * HID * 4);
    float* krel = (float*)alloc((size_t)NN * HID * 4);
    float* vrel = (float*)alloc((size_t)NN * HID * 4);
    _Float16* g16 = (_Float16*)alloc((size_t)NN * HID * 2);
    float* alpha = (float*)alloc((size_t)E * NH * 4);
    float* mbuf  = (float*)alloc((size_t)NN * NH * 4);
    float* sbuf  = (float*)alloc((size_t)NN * NH * 4);
    _Float16* wt = (_Float16*)alloc((size_t)HID * HID * 2);
    float* bf    = (float*)alloc((size_t)HID * 4);

    const dim3 B256(256);
    const int gTiles   = NN / 16;                      // 6250 GEMM row tiles
    const int gEdgeW   = (E + 7) / 8;                  // wave-per-edge blocks
    const int gNodeW   = (NN + 7) / 8;
    auto gElems = [](long n) { return (unsigned)((n + 255) / 256); };

    // ---- input projection: relu(x @ in_w + in_b), K=64 ----
    for (int t = 0; t < 2; ++t) {
        f32_to_f16<<<gElems((long)NN * 64), B256, 0, stream>>>(xin[t], xin16[t], (long)NN * 64);
        transpose_w_f16<<<gElems(64L * HID), B256, 0, stream>>>(P(IDX_INW(t)), wt, 64, HID);
        gemm_wmma_f16<<<gTiles, B256, 0, stream>>>(
            xin16[t], wt, P(IDX_INB(t)), xa32[t], xa16[t],
            nullptr, nullptr, /*act=*/1, /*K=*/64);
    }

    float* cur32[2] = { xa32[0], xa32[1] };  _Float16* cur16[2] = { xa16[0], xa16[1] };
    float* nxt32[2] = { xb32[0], xb32[1] };  _Float16* nxt16[2] = { xb16[0], xb16[1] };

    // ---- HGT layers ----
    for (int l = 0; l < 2; ++l) {
        // q projections + zero agg
        for (int t = 0; t < 2; ++t) {
            transpose_w_f16<<<gElems((long)HID * HID), B256, 0, stream>>>(P(IDX_QW(l, t)), wt, HID, HID);
            gemm_wmma_f16<<<gTiles, B256, 0, stream>>>(
                cur16[t], wt, P(IDX_QB(l, t)), qb[t], nullptr,
                nullptr, nullptr, 0, HID);
            fill_f32<<<gElems((long)NN * HID), B256, 0, stream>>>(agg[t], 0.0f, (long)NN * HID);
        }

        for (int r = 0; r < 4; ++r) {
            const int s = rsrc[r], d = rdst[r];
            // krel = x_src @ (k_w . BD(arel)) + k_b . BD(arel)   (fused GEMM)
            fuse_rel_w<<<gElems(HID * HID), B256, 0, stream>>>(
                P(IDX_KW(l, s)), P(IDX_AREL(l, r)), P(IDX_KB(l, s)), wt, bf);
            gemm_wmma_f16<<<gTiles, B256, 0, stream>>>(
                cur16[s], wt, bf, krel, nullptr, nullptr, nullptr, 0, HID);
            // vrel likewise with mrel
            fuse_rel_w<<<gElems(HID * HID), B256, 0, stream>>>(
                P(IDX_VW(l, s)), P(IDX_MREL(l, r)), P(IDX_VB(l, s)), wt, bf);
            gemm_wmma_f16<<<gTiles, B256, 0, stream>>>(
                cur16[s], wt, bf, vrel, nullptr, nullptr, nullptr, 0, HID);

            // segment softmax over incoming edges (3 passes)
            fill_f32<<<gElems((long)NN * NH), B256, 0, stream>>>(mbuf, -3.0e38f, (long)NN * NH);
            fill_f32<<<gElems((long)NN * NH), B256, 0, stream>>>(sbuf, 0.0f, (long)NN * NH);
            edge_alpha_max<<<gEdgeW, B256, 0, stream>>>(qb[d], krel, ei[r], P(IDX_PREL(l, r)),
                                                        alpha, mbuf, E);
            edge_exp_sum<<<gElems((long)E * NH), B256, 0, stream>>>(alpha, mbuf, sbuf, ei[r], E);
            edge_scatter<<<gEdgeW, B256, 0, stream>>>(vrel, alpha, sbuf, ei[r], agg[d], E);
        }

        // node update: out = sk * (gelu(agg) @ a_w + a_b) + (1-sk) * x
        for (int t = 0; t < 2; ++t) {
            gelu_to_f16<<<gElems((long)NN * HID), B256, 0, stream>>>(agg[t], g16, (long)NN * HID);
            transpose_w_f16<<<gElems((long)HID * HID), B256, 0, stream>>>(P(IDX_AW(l, t)), wt, HID, HID);
            gemm_wmma_f16<<<gTiles, B256, 0, stream>>>(
                g16, wt, P(IDX_AB(l, t)), nxt32[t], nxt16[t],
                cur32[t], P(IDX_SKIP(l, t)), 0, HID);
        }
        // swap double buffers
        for (int t = 0; t < 2; ++t) {
            float* t32 = cur32[t]; cur32[t] = nxt32[t]; nxt32[t] = t32;
            _Float16* t16 = cur16[t]; cur16[t] = nxt16[t]; nxt16[t] = t16;
        }
    }

    // ---- output head: y = x_gmd @ out_w + out_b ----
    out_proj<<<gNodeW, B256, 0, stream>>>(cur32[1], P(IDX_OUTW), P(IDX_OUTB),
                                          (float*)d_out, NN);
}
